// BiasedCrossDecoderLayer_67465346285785
// MI455X (gfx1250) — matrix operand order repaired and checked by reading, not compiled, not run on hardware
//
#include <hip/hip_runtime.h>
#include <hip/hip_bf16.h>

// ---------------------------------------------------------------------------
// Types for CDNA5 WMMA (wave32): v_wmma_f32_16x16x32_f16
// ---------------------------------------------------------------------------
typedef __attribute__((ext_vector_type(16))) _Float16 v16h;
typedef __attribute__((ext_vector_type(8)))  _Float16 v8h;
typedef __attribute__((ext_vector_type(8)))  float    v8f;

#define B_C  2
#define Q_C  1024
#define S_C  2048
#define D_C  1024
#define H_C  16
#define HD_C 64

__device__ __forceinline__ v8f wmma_f16(v16h a, v16h b, v8f c) {
    // 8 args: (neg_a, A, neg_b, B, c_mod, C, reuse_a, reuse_b)
    return __builtin_amdgcn_wmma_f32_16x16x32_f16(false, a, false, b,
                                                  (short)0, c, false, false);
}

// A-fragment 16x32 f16 (ISA 7.12.2): lane holds row M=lane%16,
// halves[0..7]  = K (g*8 .. g*8+7), halves[8..15] = K (16+g*8 ..), g=lane/16.
__device__ __forceinline__ v16h load_frag_a(const _Float16* p, int g) {
    v8h lo = *(const v8h*)(p + g * 8);
    v8h hi = *(const v8h*)(p + g * 8 + 16);
    return __builtin_shufflevector(lo, hi, 0,1,2,3,4,5,6,7,8,9,10,11,12,13,14,15);
}

// B-fragment 32x16 f16: lane holds column N=lane%16, halves[0..7] = K (g*8 ..),
// halves[8..15] = K (16+g*8 ..), g=lane/16 (same order as A).
__device__ __forceinline__ v16h load_frag_b(const _Float16* p, int g) {
    v8h lo = *(const v8h*)(p + g * 8);
    v8h hi = *(const v8h*)(p + g * 8 + 16);
    return __builtin_shufflevector(lo, hi, 0,1,2,3,4,5,6,7,8,9,10,11,12,13,14,15);
}

// ---------------------------------------------------------------------------
// LayerNorm (fp32 in) -> f16 out, one 256-thread block per row of D=1024
// ---------------------------------------------------------------------------
__global__ __launch_bounds__(256, 1)
void ln_f16_kernel(const float* __restrict__ x,
                   const float* __restrict__ gamma,
                   const float* __restrict__ beta,
                   _Float16* __restrict__ out, int D) {
    const int row = blockIdx.x;
    const int tid = threadIdx.x;
    const float* xr = x + (size_t)row * D;
    __shared__ float sm[256];
    __shared__ float sv[256];
    float s = 0.f, s2 = 0.f;
    for (int i = tid; i < D; i += 256) { float v = xr[i]; s += v; s2 += v * v; }
    sm[tid] = s; sv[tid] = s2;
    __syncthreads();
    for (int st = 128; st > 0; st >>= 1) {
        if (tid < st) { sm[tid] += sm[tid + st]; sv[tid] += sv[tid + st]; }
        __syncthreads();
    }
    const float mean = sm[0] / (float)D;
    const float var  = sv[0] / (float)D - mean * mean;
    const float inv  = rsqrtf(var + 1e-5f);
    _Float16* orow = out + (size_t)row * D;
    for (int i = tid; i < D; i += 256)
        orow[i] = (_Float16)((xr[i] - mean) * inv * gamma[i] + beta[i]);
}

// ---------------------------------------------------------------------------
// f32 -> f16 cast (weights)
// ---------------------------------------------------------------------------
__global__ __launch_bounds__(256, 1)
void cast_f16_kernel(const float* __restrict__ src,
                     _Float16* __restrict__ dst, long n) {
    long i = (long)blockIdx.x * blockDim.x + threadIdx.x;
    const long stride = (long)gridDim.x * blockDim.x;
    for (; i < n; i += stride) dst[i] = (_Float16)src[i];
}

// ---------------------------------------------------------------------------
// Generic NT GEMM:  C[M,N] = A[M,K](f16) * W[N,K](f16)^T + bias
// Block = 128 threads (4 waves).  Wave w: rows m0=(by*4+w)*16, cols
// n0=bx*128 (8 N-subtiles of 16, persistent f32 accumulators -> 1 A-load is
// reused by 8 wmma).  Both fragments load straight from global, no LDS.
// ---------------------------------------------------------------------------
constexpr int ST_RES_F32  = 0;
constexpr int ST_HEAD_F16 = 1;
constexpr int ST_VT_F16   = 2;
constexpr int ST_GELU_F16 = 3;

template <int SMODE>
__global__ __launch_bounds__(128, 1)
void gemm_nt_wmma(const _Float16* __restrict__ A,
                  const _Float16* __restrict__ W,
                  const float* __restrict__ bias,
                  const float* __restrict__ resid,
                  void* __restrict__ outp,
                  int M, int N, int K, int rowsPerBatch,
                  float outScale) {
    const int lane = threadIdx.x & 31;
    const int wave = threadIdx.x >> 5;
    const int g    = lane >> 4;
    const int ar   = lane & 15;
    const int m0   = (blockIdx.y * 4 + wave) * 16;
    const int n0   = blockIdx.x * 128;

    v8f c[8];
#pragma unroll
    for (int j = 0; j < 8; ++j) { v8f z = {}; c[j] = z; }

    const _Float16* aRow = A + (size_t)(m0 + ar) * K;
    const _Float16* bCol = W + (size_t)(n0 + ar) * K;

    for (int k0 = 0; k0 < K; k0 += 32) {
        v16h af = load_frag_a(aRow + k0, g);
#pragma unroll
        for (int j = 0; j < 8; ++j) {
            v16h bf = load_frag_b(bCol + (size_t)(j * 16) * K + k0, g);
            c[j] = wmma_f16(af, bf, c[j]);
        }
    }

#pragma unroll
    for (int j = 0; j < 8; ++j) {
        const int n = n0 + j * 16 + ar;
        const float bv = bias[n];
#pragma unroll
        for (int r = 0; r < 8; ++r) {
            const int m = m0 + r + 8 * g;
            float val = c[j][r] + bv;
            if constexpr (SMODE == ST_RES_F32) {
                float* o = (float*)outp;
                o[(size_t)m * N + n] = val + resid[(size_t)m * N + n];
            } else if constexpr (SMODE == ST_GELU_F16) {
                const float ge = 0.5f * val * (1.f + erff(val * 0.70710678f));
                ((_Float16*)outp)[(size_t)m * N + n] = (_Float16)ge;
            } else if constexpr (SMODE == ST_HEAD_F16) {
                const int bb = m / rowsPerBatch, rr = m % rowsPerBatch;
                const int hh = n >> 6, hd = n & 63;
                ((_Float16*)outp)[(((size_t)(bb * H_C + hh) * rowsPerBatch + rr) << 6) + hd] =
                    (_Float16)(val * outScale);
            } else { // ST_VT_F16
                const int bb = m / rowsPerBatch, rr = m % rowsPerBatch;
                const int hh = n >> 6, hd = n & 63;
                ((_Float16*)outp)[((size_t)(bb * H_C + hh) * 64 + hd) * rowsPerBatch + rr] =
                    (_Float16)val;
            }
        }
    }
}

// ---------------------------------------------------------------------------
// Fused flash attention, one wave (32 threads) per block.
//   block = (q-tile of 16 rows) x (one b,h);   S walked in chunks of 32.
//   scores: 2x2 wmma (K=HD=64);  row-max via 4 width-16 shuffles/row;
//   row-SUM via one extra WMMA against an all-ones B fragment (replicated
//   across lanes in accumulator layout, numerically consistent with PV);
//   P bounced through 1KB LDS (C-layout -> A-layout);  PV: 4 wmma (K=32).
//   Mask (256MB traffic floor) read once, non-temporally; next chunk's
//   K/V/mask lines prefetched with global_prefetch.
// ---------------------------------------------------------------------------
__global__ __launch_bounds__(32, 1)
void flash_attn_kernel(const _Float16* __restrict__ qh,  // (B*H,Q,64)
                       const _Float16* __restrict__ kh,  // (B*H,S,64)
                       const _Float16* __restrict__ vt,  // (B*H,64,S)
                       const float* __restrict__ mask,   // (B*H,Q,S)
                       _Float16* __restrict__ att,       // (B*Q, D)
                       int Q, int S) {
    const int lane = threadIdx.x & 31;
    const int g    = lane >> 4;
    const int nl   = lane & 15;
    const int q0   = blockIdx.x * 16;
    const int bh   = blockIdx.y;
    const int bb   = bh >> 4;   // H=16
    const int hh   = bh & 15;

    const _Float16* qb = qh + (size_t)bh * Q * 64;
    const _Float16* kb = kh + (size_t)bh * S * 64;
    const _Float16* vb = vt + (size_t)bh * 64 * S;
    const float*    mb = mask + ((size_t)bh * Q + q0) * S;

    __shared__ _Float16 ldsP[16 * 32];

    v16h qa[2];
    {
        const _Float16* p = qb + (size_t)(q0 + nl) * 64;
#pragma unroll
        for (int ks = 0; ks < 2; ++ks) qa[ks] = load_frag_a(p + ks * 32, g);
    }
    v16h ones;
#pragma unroll
    for (int i = 0; i < 16; ++i) ones[i] = (_Float16)1.f;

    v8f acc[4];
#pragma unroll
    for (int j = 0; j < 4; ++j) { v8f z = {}; acc[j] = z; }
    float mrow[8], lrow[8];
#pragma unroll
    for (int r = 0; r < 8; ++r) { mrow[r] = -3.0e38f; lrow[r] = 0.f; }

    for (int s0 = 0; s0 < S; s0 += 32) {
        // prefetch next chunk (K rows / V^T lines / mask rows) into caches
        if (s0 + 32 < S) {
            __builtin_prefetch(kb + (size_t)(s0 + 32 + lane) * 64, 0, 1);
            __builtin_prefetch(vb + (size_t)(2 * lane) * S + s0 + 32, 0, 1);
            __builtin_prefetch(vb + (size_t)(2 * lane + 1) * S + s0 + 32, 0, 1);
            __builtin_prefetch(mb + (size_t)nl * S + s0 + 32, 0, 1);
        }
        // ---- scores: 16x32 tile of q.k^T (q already scaled) ----
        v8f sc[2];
#pragma unroll
        for (int nc = 0; nc < 2; ++nc) {
            v8f z = {}; sc[nc] = z;
#pragma unroll
            for (int ks = 0; ks < 2; ++ks) {
                v16h bf = load_frag_b(kb + (size_t)(s0 + nc * 16 + nl) * 64 + ks * 32, g);
                sc[nc] = wmma_f16(qa[ks], bf, sc[nc]);
            }
        }
        // ---- mask add + online max update (C layout: M=r+8g, N=nl) ----
        float alpha8[8];
#pragma unroll
        for (int r = 0; r < 8; ++r) {
            const int mloc = r + 8 * g;
            const float* mp = mb + (size_t)mloc * S + s0 + nl;
            const float v0 = sc[0][r] + __builtin_nontemporal_load(mp);
            const float v1 = sc[1][r] + __builtin_nontemporal_load(mp + 16);
            float t = fmaxf(v0, v1);
#pragma unroll
            for (int off = 1; off < 16; off <<= 1) t = fmaxf(t, __shfl_xor(t, off, 16));
            const float nm = fmaxf(mrow[r], t);
            alpha8[r] = __expf(mrow[r] - nm);
            mrow[r]   = nm;
            ldsP[mloc * 32 + nl]      = (_Float16)__expf(v0 - nm);
            ldsP[mloc * 32 + 16 + nl] = (_Float16)__expf(v1 - nm);
        }
        __syncthreads();
        // ---- P as A-frag; row-sums via WMMA against ones; then P x V ----
        v16h pa = load_frag_a(&ldsP[(size_t)nl * 32], g);
        v8f zz = {};
        v8f rsum = wmma_f16(pa, ones, zz);   // rsum[r] = sum_k P[m,k], all lanes
#pragma unroll
        for (int r = 0; r < 8; ++r) {
            lrow[r] = lrow[r] * alpha8[r] + rsum[r];
#pragma unroll
            for (int j = 0; j < 4; ++j) acc[j][r] *= alpha8[r];
        }
#pragma unroll
        for (int j = 0; j < 4; ++j) {
            v16h bf = load_frag_b(vb + (size_t)(j * 16 + nl) * S + s0, g);
            acc[j] = wmma_f16(pa, bf, acc[j]);
        }
        __syncthreads();
    }

    // ---- normalize and write att as (B*Q, D) f16 for the out-proj GEMM ----
#pragma unroll
    for (int j = 0; j < 4; ++j) {
#pragma unroll
        for (int r = 0; r < 8; ++r) {
            const int mloc = r + 8 * g;
            const int qq   = q0 + mloc;
            const float val = acc[j][r] / lrow[r];
            att[(size_t)(bb * Q + qq) * D_C + hh * 64 + j * 16 + nl] = (_Float16)val;
        }
    }
}

// ---------------------------------------------------------------------------
// Host-side orchestration
// ---------------------------------------------------------------------------
extern "C" void kernel_launch(void* const* d_in, const int* in_sizes, int n_in,
                              void* d_out, int out_size, void* d_ws, size_t ws_size,
                              hipStream_t stream) {
    (void)in_sizes; (void)n_in; (void)out_size; (void)ws_size;
    const int B = B_C, Q = Q_C, S = S_C, D = D_C, H = H_C;

    const float* x      = (const float*)d_in[0];
    const float* z      = (const float*)d_in[1];
    const float* mask   = (const float*)d_in[2];
    const float* gq     = (const float*)d_in[3];
    const float* bq     = (const float*)d_in[4];
    const float* gkv    = (const float*)d_in[5];
    const float* bkv    = (const float*)d_in[6];
    const float* gff    = (const float*)d_in[7];
    const float* bff    = (const float*)d_in[8];
    const float* in_w   = (const float*)d_in[9];
    const float* in_b   = (const float*)d_in[10];
    const float* out_w  = (const float*)d_in[11];
    const float* out_b  = (const float*)d_in[12];
    const float* w1     = (const float*)d_in[13];
    const float* b1     = (const float*)d_in[14];
    const float* w2     = (const float*)d_in[15];
    const float* b2     = (const float*)d_in[16];
    float* outp = (float*)d_out;

    char* ws = (char*)d_ws;
    auto alloc = [&](size_t bytes) -> void* {
        void* p = ws;
        ws += (bytes + 255) & ~(size_t)255;
        return p;
    };
    _Float16* xn_h   = (_Float16*)alloc((size_t)B * Q * D * 2);
    _Float16* zn_h   = (_Float16*)alloc((size_t)B * S * D * 2);
    _Float16* win_h  = (_Float16*)alloc((size_t)3 * D * D * 2);
    _Float16* wout_h = (_Float16*)alloc((size_t)D * D * 2);
    _Float16* w1_h   = (_Float16*)alloc((size_t)4 * D * D * 2);
    _Float16* w2_h   = (_Float16*)alloc((size_t)4 * D * D * 2);
    _Float16* q_h    = (_Float16*)alloc((size_t)B * Q * D * 2);
    _Float16* k_h    = (_Float16*)alloc((size_t)B * S * D * 2);
    _Float16* v_t    = (_Float16*)alloc((size_t)B * S * D * 2);
    _Float16* att_h  = (_Float16*)alloc((size_t)B * Q * D * 2);
    float*    x_res  = (float*)   alloc((size_t)B * Q * D * 4);
    _Float16* h_h    = (_Float16*)alloc((size_t)B * Q * D * 2);
    _Float16* ff1_h  = (_Float16*)alloc((size_t)B * Q * 4 * D * 2);

    // 1) LayerNorms (fp32 -> f16)
    ln_f16_kernel<<<B * Q, 256, 0, stream>>>(x, gq, bq, xn_h, D);
    ln_f16_kernel<<<B * S, 256, 0, stream>>>(z, gkv, bkv, zn_h, D);

    // 2) weight casts (one-time, stays hot in 192MB L2)
    cast_f16_kernel<<<1024, 256, 0, stream>>>(in_w,  win_h,  (long)3 * D * D);
    cast_f16_kernel<<<1024, 256, 0, stream>>>(out_w, wout_h, (long)D * D);
    cast_f16_kernel<<<1024, 256, 0, stream>>>(w1,    w1_h,   (long)4 * D * D);
    cast_f16_kernel<<<1024, 256, 0, stream>>>(w2,    w2_h,   (long)4 * D * D);

    // 3) QKV projections (NT GEMM).  q scaled by 1/sqrt(64); v stored (B*H,64,S).
    gemm_nt_wmma<ST_HEAD_F16><<<dim3(D / 128, (B * Q) / 64), 128, 0, stream>>>(
        xn_h, win_h, in_b, nullptr, q_h, B * Q, D, D, Q, 0.125f);
    gemm_nt_wmma<ST_HEAD_F16><<<dim3(D / 128, (B * S) / 64), 128, 0, stream>>>(
        zn_h, win_h + (size_t)D * D, in_b + D, nullptr, k_h, B * S, D, D, S, 1.0f);
    gemm_nt_wmma<ST_VT_F16><<<dim3(D / 128, (B * S) / 64), 128, 0, stream>>>(
        zn_h, win_h + (size_t)2 * D * D, in_b + 2 * D, nullptr, v_t, B * S, D, D, S, 1.0f);

    // 4) fused flash attention (mask read exactly once; no logits/probs in HBM)
    flash_attn_kernel<<<dim3(Q / 16, B * H), 32, 0, stream>>>(
        q_h, k_h, v_t, mask, att_h, Q, S);

    // 5) out-projection + residual:  x_res = x + att @ out_w^T + out_b
    gemm_nt_wmma<ST_RES_F32><<<dim3(D / 128, (B * Q) / 64), 128, 0, stream>>>(
        att_h, wout_h, out_b, x, x_res, B * Q, D, D, 0, 1.0f);

    // 6) FFN
    ln_f16_kernel<<<B * Q, 256, 0, stream>>>(x_res, gff, bff, h_h, D);
    gemm_nt_wmma<ST_GELU_F16><<<dim3((4 * D) / 128, (B * Q) / 64), 128, 0, stream>>>(
        h_h, w1_h, b1, nullptr, ff1_h, B * Q, 4 * D, D, 0, 1.0f);
    gemm_nt_wmma<ST_RES_F32><<<dim3(D / 128, (B * Q) / 64), 128, 0, stream>>>(
        ff1_h, w2_h, b2, x_res, outp, B * Q, D, 4 * D, 0, 1.0f);
}
